// LabelwiseAttention_34557306863626
// MI455X (gfx1250) — matrix operand order repaired and checked by reading, not compiled, not run on hardware
//
#include <hip/hip_runtime.h>

typedef __attribute__((ext_vector_type(16))) __bf16 v16bf;
typedef __attribute__((ext_vector_type(8)))  float  v8f;

#define B_   8
#define S_   2048
#define D_   768
#define C_   4271
#define CP_  4288            // C padded to multiple of 64
#define NW_  4               // waves per block (wave32)
#define TPB  (NW_ * 32)      // 128 threads
#define SCH  32              // s-rows per chunk (two 16-row WMMA M tiles)

// ---------------- bf16 conversion (RNE) ----------------
__device__ __forceinline__ unsigned short f32_to_bf16(float f) {
    unsigned u = __float_as_uint(f);
    u += 0x7FFFu + ((u >> 16) & 1u);
    return (unsigned short)(u >> 16);
}

__global__ void cvt_x_kernel(const float* __restrict__ src,
                             unsigned short* __restrict__ dst, long n) {
    long i = (long)blockIdx.x * blockDim.x + threadIdx.x;
    if (i < n) dst[i] = f32_to_bf16(src[i]);
}

__global__ void cvt_w_kernel(const float* __restrict__ src,
                             unsigned short* __restrict__ dst,
                             long nsrc, long ndst) {
    long i = (long)blockIdx.x * blockDim.x + threadIdx.x;
    if (i < ndst) dst[i] = (i < nsrc) ? f32_to_bf16(src[i]) : (unsigned short)0;
}

// ---------------- async global -> LDS staging (gfx1250) ----------------
// Copies n16 uint4's (16B per lane per issue) from gsrc to LDS dst.
// Tracked on ASYNCcnt; caller must s_wait_asynccnt + barrier before reading.
template <int N16>
__device__ __forceinline__ void stage_async(unsigned short* dstLds,
                                            const unsigned short* gsrc,
                                            int tid) {
    const unsigned lds_base = (unsigned)(uintptr_t)dstLds;   // low 32 bits = LDS offset
    const unsigned long long g_base = (unsigned long long)(uintptr_t)gsrc;
    #pragma unroll
    for (int i = 0; i < N16 / TPB; ++i) {
        const int idx = (tid + i * TPB) * 16;                // byte offset
        unsigned           lds = lds_base + (unsigned)idx;
        unsigned long long ga  = g_base + (unsigned long long)idx;
        asm volatile("global_load_async_to_lds_b128 %0, %1, off"
                     :: "v"(lds), "v"(ga) : "memory");
    }
}

__device__ __forceinline__ void wait_async_all() {
    asm volatile("s_wait_asynccnt 0" ::: "memory");
}

// ---------------- fused labelwise-attention kernel ----------------
union Frag { uint4 q[2]; v16bf v; };

__global__ void __launch_bounds__(TPB)
lwattn_kernel(const unsigned short* __restrict__ xb,   // [B][S][D] bf16
              const unsigned short* __restrict__ wab,  // [CP][D] bf16
              const unsigned short* __restrict__ wob,  // [CP][D] bf16
              const float* __restrict__ bo,            // [C]
              float* __restrict__ out)                 // [B][C]
{
    extern __shared__ unsigned short smem[];
    unsigned short* xs0 = smem;                  // 32 x 768 bf16 (48 KB)
    unsigned short* xs1 = xs0 + SCH * D_;        // 32 x 768 bf16 (48 KB)
    unsigned short* swa = xs1 + SCH * D_;        // 64 x 768 bf16 (96 KB)
    unsigned short* swo = swa + 64 * D_;         // 64 x 768 bf16 (96 KB)

    const int tid   = threadIdx.x;
    const int wave  = tid >> 5;
    const int lane  = tid & 31;
    const int laneN = lane & 15;           // class-within-wave (WMMA N)
    const int hi    = lane >> 4;           // lane half selects K sub-range

    const int b  = blockIdx.y;
    const int c0 = blockIdx.x * 64;

    const unsigned short* gx_base = xb + (size_t)b * S_ * D_;

    // Stage W tiles (once) and x chunk 0 via async DMA to LDS
    stage_async<64 * D_ / 8>(swa, wab + (size_t)c0 * D_, tid);
    stage_async<64 * D_ / 8>(swo, wob + (size_t)c0 * D_, tid);
    stage_async<SCH * D_ / 8>(xs0, gx_base, tid);
    wait_async_all();
    __syncthreads();

    const uint4* swa4 = (const uint4*)swa;
    const uint4* swo4 = (const uint4*)swo;

    // online-softmax running state, replicated in lane pairs (c, c+16)
    float m = -INFINITY, l = 0.f, acc = 0.f;

    const int arow = laneN * (D_ / 8);                 // A: row M = laneN (tile 0)
    const int wrow = (wave * 16 + laneN) * (D_ / 8);   // B: class row

    for (int it = 0; it < S_ / SCH; ++it) {
        const uint4* c4 = (const uint4*)((it & 1) ? xs1 : xs0);
        unsigned short* nxt = (it & 1) ? xs0 : xs1;

        // kick off next chunk's DMA; overlaps with the WMMA loop below
        if (it + 1 < S_ / SCH)
            stage_async<SCH * D_ / 8>(nxt, gx_base + (size_t)(it + 1) * SCH * D_, tid);

        v8f accA0 = {0.f,0.f,0.f,0.f,0.f,0.f,0.f,0.f};
        v8f accA1 = {0.f,0.f,0.f,0.f,0.f,0.f,0.f,0.f};
        v8f accO0 = {0.f,0.f,0.f,0.f,0.f,0.f,0.f,0.f};
        v8f accO1 = {0.f,0.f,0.f,0.f,0.f,0.f,0.f,0.f};

        #pragma unroll
        for (int k0 = 0; k0 < D_; k0 += 32) {
            Frag fa0, fa1, fwa, fwo;
            // A 16x32 bf16: lanes0-15 K=0..15, lanes16-31 K=16..31 (contiguous 32B)
            const int a0 = arow + (k0 >> 3) + hi * 2;
            fa0.q[0] = c4[a0];
            fa0.q[1] = c4[a0 + 1];
            const int a1 = a0 + 16 * (D_ / 8);         // rows 16..31
            fa1.q[0] = c4[a1];
            fa1.q[1] = c4[a1 + 1];
            // B 32x16 bf16: lanes0-15 K=0..15, lanes16-31 K=16..31 (contiguous 32B)
            const int wi = wrow + (k0 >> 3) + hi * 2;
            fwa.q[0] = swa4[wi];
            fwa.q[1] = swa4[wi + 1];
            fwo.q[0] = swo4[wi];
            fwo.q[1] = swo4[wi + 1];
            accA0 = __builtin_amdgcn_wmma_f32_16x16x32_bf16(
                        false, fa0.v, false, fwa.v, (short)0, accA0, false, false);
            accA1 = __builtin_amdgcn_wmma_f32_16x16x32_bf16(
                        false, fa1.v, false, fwa.v, (short)0, accA1, false, false);
            accO0 = __builtin_amdgcn_wmma_f32_16x16x32_bf16(
                        false, fa0.v, false, fwo.v, (short)0, accO0, false, false);
            accO1 = __builtin_amdgcn_wmma_f32_16x16x32_bf16(
                        false, fa1.v, false, fwo.v, (short)0, accO1, false, false);
        }

        // ---- online softmax update over this 32-row s chunk ----
        float cmax = accA0[0];
        #pragma unroll
        for (int r = 1; r < 8; ++r) cmax = fmaxf(cmax, accA0[r]);
        #pragma unroll
        for (int r = 0; r < 8; ++r) cmax = fmaxf(cmax, accA1[r]);
        cmax = fmaxf(cmax, __shfl_xor(cmax, 16));          // close column (32 rows)
        const float nm   = fmaxf(m, cmax);
        const float corr = __expf(m - nm);
        float psum = 0.f, pdot = 0.f;
        #pragma unroll
        for (int r = 0; r < 8; ++r) {
            const float p0 = __expf(accA0[r] - nm);
            const float p1 = __expf(accA1[r] - nm);
            psum += p0 + p1;
            pdot += p0 * accO0[r] + p1 * accO1[r];
        }
        psum += __shfl_xor(psum, 16);
        pdot += __shfl_xor(pdot, 16);
        l   = l * corr + psum;
        acc = acc * corr + pdot;
        m   = nm;

        // next chunk's DMA must have landed; all waves done reading cur
        wait_async_all();
        __syncthreads();
    }

    if (hi == 0) {
        const int c = c0 + wave * 16 + laneN;
        if (c < C_) {
            const float logit = acc / l + bo[c];
            out[(size_t)b * C_ + c] = 1.0f / (1.0f + __expf(-logit));
        }
    }
}

// ---------------- host launch ----------------
extern "C" void kernel_launch(void* const* d_in, const int* in_sizes, int n_in,
                              void* d_out, int out_size, void* d_ws, size_t ws_size,
                              hipStream_t stream) {
    (void)in_sizes; (void)n_in; (void)out_size; (void)ws_size;
    const float* x  = (const float*)d_in[0];
    const float* Wa = (const float*)d_in[1];
    const float* Wo = (const float*)d_in[2];
    const float* bo = (const float*)d_in[3];
    float* out = (float*)d_out;

    unsigned short* xb  = (unsigned short*)d_ws;                 // B*S*D bf16
    unsigned short* wab = xb  + (size_t)B_ * S_ * D_;            // CP*D bf16
    unsigned short* wob = wab + (size_t)CP_ * D_;                // CP*D bf16

    const long nx = (long)B_ * S_ * D_;
    const long nw_src = (long)C_ * D_;
    const long nw_dst = (long)CP_ * D_;

    cvt_x_kernel<<<(int)((nx + 255) / 256), 256, 0, stream>>>(x, xb, nx);
    cvt_w_kernel<<<(int)((nw_dst + 255) / 256), 256, 0, stream>>>(Wa, wab, nw_src, nw_dst);
    cvt_w_kernel<<<(int)((nw_dst + 255) / 256), 256, 0, stream>>>(Wo, wob, nw_src, nw_dst);

    const size_t smem_bytes = (size_t)(2 * SCH * D_ + 2 * 64 * D_) * sizeof(unsigned short); // 288 KB
    (void)hipFuncSetAttribute((const void*)lwattn_kernel,
                              hipFuncAttributeMaxDynamicSharedMemorySize,
                              (int)smem_bytes);

    dim3 grid(CP_ / 64, B_);   // 67 class tiles x 8 batches
    lwattn_kernel<<<grid, TPB, smem_bytes, stream>>>(xb, wab, wob, bo, out);
}
